// GemNetTDecoder_86234353369157
// MI455X (gfx1250) — matrix operand order repaired and checked by reading, not compiled, not run on hardware
//
#include <hip/hip_runtime.h>
#include <hip/hip_bf16.h>
#include <math.h>

typedef __attribute__((ext_vector_type(16))) _Float16 v16h;
typedef __attribute__((ext_vector_type(8)))  _Float16 v8h;
typedef __attribute__((ext_vector_type(4)))  _Float16 v4h;
typedef __attribute__((ext_vector_type(8)))  float    v8f;

#define HID     512
#define NFRQ    128
#define DIS_DIM 768           // NFRQ*2*3
#define NGRAPH  32
#define NPG     24
#define NNODES  768           // NGRAPH*NPG
#define NEDGES  18432         // NGRAPH*NPG*NPG
#define MAXA    100
#define MAXA_P  128           // padded fc_atom output width
#define NLAYERS 6

__device__ __forceinline__ float silu_f(float x) { return x / (1.0f + __expf(-x)); }

// ---------------------------------------------------------------------------
// WMMA GEMM: C[M,N] = act(A[M,K] @ W[K,N] + bias) (+ resid), f16 math f32 acc
// Tile: 128x64x32. 8 waves/block, each wave owns a 32x32 subtile (2x2 WMMA).
// Register double-buffered: while tile t is consumed from LDS by the WMMAs,
// tile t+1's global loads are in flight into registers. LDS layouts give
// 16B-aligned ds_load_b128 fragment reads matching the ISA A/B VGPR layouts.
// REQUIRES: M % 128 == 0, K % 32 == 0, N % 64 == 0  -> fully branch-free.
// ---------------------------------------------------------------------------
#define BM 128
#define BN 64
#define BK 32
#define ASTR 40   // half-element row stride (80B rows -> aligned subchunks)
#define BSTR 40

template <int A_HALF>
__global__ __launch_bounds__(256) void wmma_gemm_kernel(
    const void* __restrict__ Av,
    const float* __restrict__ W, const float* __restrict__ bias,
    const float* __restrict__ resid, float* __restrict__ C,
    int M, int N, int K, int ldc, int act_silu)
{
  __shared__ _Float16 As[BM][ASTR];
  __shared__ _Float16 Bst[BN][BSTR];   // transposed: Bst[n][k]

  const int tid  = threadIdx.x;
  const int m0   = blockIdx.y * BM;
  const int n0   = blockIdx.x * BN;
  const int wave = tid >> 5, lane = tid & 31;
  const int wm   = wave & 3, wn = wave >> 2;       // 4 waves along M, 2 along N
  const int l16  = lane & 15, hb = lane >> 4;      // half-wave select

  const float*    Af = (const float*)Av;
  const _Float16* Ah = (const _Float16*)Av;

  // ---- hoisted, k0-invariant staging indices ----
  // A (f32 path): 4 chunks of float4; A (f16 path): 2 chunks of v8h
  int    a_r[4], a_c[4];
  size_t a_base[4];
  #pragma unroll
  for (int i = 0; i < 4; ++i) {
    int v = i * 256 + tid;
    if (A_HALF) { a_r[i] = v >> 2; a_c[i] = (v & 3) * 8; }
    else        { a_r[i] = v >> 3; a_c[i] = (v & 7) * 4; }
    a_base[i] = (size_t)(m0 + a_r[i]) * K + a_c[i];
  }
  // W: 2 chunks of float4, scattered transposed
  int    w_kr[2], w_nc[2];
  size_t w_base[2];
  #pragma unroll
  for (int i = 0; i < 2; ++i) {
    int v = i * 256 + tid;
    w_kr[i] = v >> 4; w_nc[i] = (v & 15) * 4;
    w_base[i] = (size_t)w_kr[i] * N + n0 + w_nc[i];
  }

  // ---- register buffers for the in-flight tile ----
  float4 aw[4]; v8h awh[2]; float4 ww[2];

  // initial load (k0 = 0)
  if (A_HALF) {
    #pragma unroll
    for (int i = 0; i < 2; ++i) awh[i] = *(const v8h*)(Ah + a_base[i]);
  } else {
    #pragma unroll
    for (int i = 0; i < 4; ++i) aw[i] = *(const float4*)(Af + a_base[i]);
  }
  #pragma unroll
  for (int i = 0; i < 2; ++i) ww[i] = *(const float4*)(W + w_base[i]);

  v8f acc[2][2] = {};

  for (int k0 = 0; k0 < K; k0 += BK) {
    // ---- dump the loaded register tile into LDS
    if (A_HALF) {
      #pragma unroll
      for (int i = 0; i < 2; ++i) *(v8h*)&As[a_r[i]][a_c[i]] = awh[i];
    } else {
      #pragma unroll
      for (int i = 0; i < 4; ++i) {
        v4h hv;
        hv[0] = (_Float16)aw[i].x; hv[1] = (_Float16)aw[i].y;
        hv[2] = (_Float16)aw[i].z; hv[3] = (_Float16)aw[i].w;
        *(v4h*)&As[a_r[i]][a_c[i]] = hv;
      }
    }
    #pragma unroll
    for (int i = 0; i < 2; ++i) {
      Bst[w_nc[i] + 0][w_kr[i]] = (_Float16)ww[i].x;
      Bst[w_nc[i] + 1][w_kr[i]] = (_Float16)ww[i].y;
      Bst[w_nc[i] + 2][w_kr[i]] = (_Float16)ww[i].z;
      Bst[w_nc[i] + 3][w_kr[i]] = (_Float16)ww[i].w;
    }
    __syncthreads();

    // ---- issue next tile's global loads (overlap with WMMA below)
    const int kn = k0 + BK;
    if (kn < K) {
      if (A_HALF) {
        #pragma unroll
        for (int i = 0; i < 2; ++i) awh[i] = *(const v8h*)(Ah + a_base[i] + kn);
      } else {
        #pragma unroll
        for (int i = 0; i < 4; ++i) aw[i] = *(const float4*)(Af + a_base[i] + kn);
      }
      #pragma unroll
      for (int i = 0; i < 2; ++i) ww[i] = *(const float4*)(W + w_base[i] + (size_t)kn * N);
    }

    // ---- build fragments per ISA layouts, 16B-aligned LDS reads
    v16h afr[2], bfr[2];
    #pragma unroll
    for (int t = 0; t < 2; ++t) {
      int row = wm * 32 + t * 16 + l16;            // A: lane -> M row
      v8h alo = *(const v8h*)&As[row][hb * 8];     // K = hb*8 .. +7
      v8h ahi = *(const v8h*)&As[row][16 + hb * 8];// K = 16+hb*8 .. +7
      afr[t] = __builtin_shufflevector(alo, ahi, 0,1,2,3,4,5,6,7,8,9,10,11,12,13,14,15);
      int col = wn * 32 + t * 16 + l16;            // B: lane -> N col
      v8h blo = *(const v8h*)&Bst[col][hb * 8];        // K = hb*8 .. +7
      v8h bhi = *(const v8h*)&Bst[col][16 + hb * 8];   // K = 16+hb*8 .. +7
      bfr[t] = __builtin_shufflevector(blo, bhi, 0,1,2,3,4,5,6,7,8,9,10,11,12,13,14,15);
    }
    #pragma unroll
    for (int tm = 0; tm < 2; ++tm)
      #pragma unroll
      for (int tn = 0; tn < 2; ++tn)
        acc[tm][tn] = __builtin_amdgcn_wmma_f32_16x16x32_f16(
            false, afr[tm], false, bfr[tn], (short)0, acc[tm][tn], false, false);
    __syncthreads();
  }

  // ---- epilogue: bias -> silu -> residual -> store (C VGPR layout per ISA)
  #pragma unroll
  for (int tm = 0; tm < 2; ++tm) {
    #pragma unroll
    for (int tn = 0; tn < 2; ++tn) {
      int colg = n0 + wn * 32 + tn * 16 + l16;
      float bv = bias ? bias[colg] : 0.0f;
      #pragma unroll
      for (int r = 0; r < 8; ++r) {
        int rowg = m0 + wm * 32 + tm * 16 + r + hb * 8;
        float v = acc[tm][tn][r] + bv;
        if (act_silu) v = silu_f(v);
        if (resid) v += resid[(size_t)rowg * ldc + colg];
        C[(size_t)rowg * ldc + colg] = v;
      }
    }
  }
}

// ---------------------------------------------------------------------------
// Sinusoidal distance features, stored f16 (GEMM A operand only).
// dis[e, c]: c<384 -> sin, else cos; within half: dim = c/128, f = c%128.
// ---------------------------------------------------------------------------
__global__ __launch_bounds__(256) void k_dis(const float* __restrict__ frac,
                                             const int* __restrict__ esrc,
                                             const int* __restrict__ edst,
                                             _Float16* __restrict__ dis)
{
  int idx = blockIdx.x * 256 + threadIdx.x;           // NEDGES*768 exact
  int e = idx / DIS_DIM, c = idx % DIS_DIM;
  int s = esrc[e], d = edst[e];
  int half = (c >= 384);
  int cc = half ? c - 384 : c;
  int dim = cc / NFRQ, f = cc % NFRQ;
  float fd = frac[d * 3 + dim] - frac[s * 3 + dim];
  fd -= floorf(fd);                                    // mod 1.0
  float ang = 6.28318530717958647f * (float)f * fd;
  float v = half ? __cosf(ang) : __sinf(ang);
  dis[idx] = (_Float16)v;
}

// lattice inner products: lat_ip[g, i*3+k] = <L[g,i,:], L[g,k,:]>
__global__ __launch_bounds__(256) void k_latip(const float* __restrict__ L,
                                               float* __restrict__ latip)
{
  int idx = blockIdx.x * 256 + threadIdx.x;
  if (idx >= NGRAPH * 9) return;
  int g = idx / 9, i = (idx % 9) / 3, k = idx % 3;
  float s = 0.f;
  #pragma unroll
  for (int j = 0; j < 3; ++j) s += L[g * 9 + i * 3 + j] * L[g * 9 + k * 3 + j];
  latip[idx] = s;
}

// node embedding: t0 = atom_types[768,100] @ Wne[100,512] + bne
__global__ __launch_bounds__(256) void k_h0(const float* __restrict__ at,
                                            const float* __restrict__ Wne,
                                            const float* __restrict__ bne,
                                            float* __restrict__ t0)
{
  int idx = blockIdx.x * 256 + threadIdx.x;           // NNODES*HID exact
  int n = idx / HID, j = idx % HID;
  float s = bne[j];
  for (int a = 0; a < MAXA; ++a) s += at[n * MAXA + a] * Wne[a * HID + j];
  t0[idx] = s;
}

// LayerNorm over HID=512, one block per row
__global__ __launch_bounds__(256) void k_ln(const float* __restrict__ x,
                                            const float* __restrict__ g,
                                            const float* __restrict__ b,
                                            float* __restrict__ y)
{
  __shared__ float s1[256], s2[256];
  int row = blockIdx.x, t = threadIdx.x;
  float x0 = x[row * HID + t], x1 = x[row * HID + 256 + t];
  s1[t] = x0 + x1; s2[t] = x0 * x0 + x1 * x1;
  __syncthreads();
  for (int off = 128; off > 0; off >>= 1) {
    if (t < off) { s1[t] += s1[t + off]; s2[t] += s2[t + off]; }
    __syncthreads();
  }
  float mean = s1[0] * (1.0f / HID);
  float var  = s2[0] * (1.0f / HID) - mean * mean;
  float rs   = rsqrtf(var + 1e-5f);
  y[row * HID + t]       = (x0 - mean) * rs * g[t] + b[t];
  y[row * HID + 256 + t] = (x1 - mean) * rs * g[256 + t] + b[256 + t];
}

// Q = lat_ip[32,9] @ W_lat[9,512]   (W_lat = edge1.W rows 1024..1032)
__global__ __launch_bounds__(256) void k_qlat(const float* __restrict__ latip,
                                              const float* __restrict__ Wlat,
                                              float* __restrict__ Q)
{
  int idx = blockIdx.x * 256 + threadIdx.x;           // 32*512 exact
  int g = idx / HID, j = idx % HID;
  float s = 0.f;
  #pragma unroll
  for (int i = 0; i < 9; ++i) s += latip[g * 9 + i] * Wlat[i * HID + j];
  Q[idx] = s;
}

// ef1 = silu(R + Psrc[src] + Pdst[dst] + Q[graph] + b1), in place over R
__global__ __launch_bounds__(256) void k_combine(float* __restrict__ R,
                                                 const float* __restrict__ P,
                                                 const float* __restrict__ Q,
                                                 const float* __restrict__ b1,
                                                 const int* __restrict__ esrc,
                                                 const int* __restrict__ edst,
                                                 const int* __restrict__ n2g)
{
  int idx = blockIdx.x * 256 + threadIdx.x;           // NEDGES*HID exact
  int e = idx >> 9, j = idx & (HID - 1);
  int s = esrc[e], d = edst[e], g = n2g[s];
  float v = R[idx]
          + P[(size_t)s * 1024 + j]
          + P[(size_t)d * 1024 + 512 + j]
          + Q[g * HID + j] + b1[j];
  R[idx] = silu_f(v);
}

// agg[n] = mean over the 24 contiguous edges owned by node n (edges sorted by src)
__global__ __launch_bounds__(256) void k_agg(const float* __restrict__ ef2,
                                             float* __restrict__ agg)
{
  int n = blockIdx.x, t = threadIdx.x;
  const float* base = ef2 + (size_t)n * NPG * HID;
  float s0 = 0.f, s1 = 0.f;
  #pragma unroll 4
  for (int d = 0; d < NPG; ++d) { s0 += base[d * HID + t]; s1 += base[d * HID + 256 + t]; }
  agg[n * HID + t]       = s0 * (1.0f / NPG);
  agg[n * HID + 256 + t] = s1 * (1.0f / NPG);
}

// ncat = [hl | agg]  (768 x 1024)
__global__ __launch_bounds__(256) void k_concat(const float* __restrict__ hl,
                                                const float* __restrict__ agg,
                                                float* __restrict__ ncat)
{
  int idx = blockIdx.x * 256 + threadIdx.x;           // NNODES*1024 exact
  int n = idx >> 10, c = idx & 1023;
  ncat[idx] = (c < HID) ? hl[n * HID + c] : agg[n * HID + (c - HID)];
}

// coord_out[n,c] = <hF[n,:], Wc[:,c]>
__global__ __launch_bounds__(256) void k_coord(const float* __restrict__ hF,
                                               const float* __restrict__ Wc,
                                               float* __restrict__ out)
{
  int idx = blockIdx.x * 256 + threadIdx.x;           // NNODES*3 exact
  int n = idx / 3, c = idx % 3;
  float s = 0.f;
  for (int j = 0; j < HID; ++j) s += hF[n * HID + j] * Wc[j * 3 + c];
  out[idx] = s;
}

// g_feat[g] = mean over 24 nodes of hF
__global__ __launch_bounds__(256) void k_gfeat(const float* __restrict__ hF,
                                               float* __restrict__ gf)
{
  int g = blockIdx.x, t = threadIdx.x;
  const float* base = hF + (size_t)g * NPG * HID;
  float s0 = 0.f, s1 = 0.f;
  #pragma unroll 4
  for (int i = 0; i < NPG; ++i) { s0 += base[i * HID + t]; s1 += base[i * HID + 256 + t]; }
  gf[g * HID + t]       = s0 * (1.0f / NPG);
  gf[g * HID + 256 + t] = s1 * (1.0f / NPG);
}

// latm[g,c] = <gfeat[g,:], Wl[:,c]>  (c < 9)
__global__ __launch_bounds__(256) void k_lat1(const float* __restrict__ gf,
                                              const float* __restrict__ Wl,
                                              float* __restrict__ latm)
{
  int idx = blockIdx.x * 256 + threadIdx.x;
  if (idx >= NGRAPH * 9) return;
  int g = idx / 9, c = idx % 9;
  float s = 0.f;
  for (int j = 0; j < HID; ++j) s += gf[g * HID + j] * Wl[j * 9 + c];
  latm[idx] = s;
}

// lattice_out[g] = latm[g](3x3) @ input_lattice[g](3x3)
__global__ __launch_bounds__(256) void k_lat2(const float* __restrict__ latm,
                                              const float* __restrict__ L,
                                              float* __restrict__ out)
{
  int idx = blockIdx.x * 256 + threadIdx.x;
  if (idx >= NGRAPH * 9) return;
  int g = idx / 9, i = (idx % 9) / 3, k = idx % 3;
  float s = 0.f;
  #pragma unroll
  for (int j = 0; j < 3; ++j) s += latm[g * 9 + i * 3 + j] * L[g * 9 + j * 3 + k];
  out[idx] = s;
}

// zero-pad fc_atom weight [512,100] -> [512,128] and bias [100] -> [128]
__global__ __launch_bounds__(256) void k_padfc(const float* __restrict__ W,
                                               const float* __restrict__ b,
                                               float* __restrict__ Wp,
                                               float* __restrict__ bp)
{
  int idx = blockIdx.x * 256 + threadIdx.x;           // HID*MAXA_P exact
  int k = idx >> 7, c = idx & (MAXA_P - 1);
  Wp[idx] = (c < MAXA) ? W[k * MAXA + c] : 0.0f;
  if (idx < MAXA_P) bp[idx] = (idx < MAXA) ? b[idx] : 0.0f;
}

// extract the 100 valid columns of the padded fc_atom result
__global__ __launch_bounds__(256) void k_extract(const float* __restrict__ Cp,
                                                 float* __restrict__ out)
{
  int idx = blockIdx.x * 256 + threadIdx.x;           // NNODES*MAXA exact
  int n = idx / MAXA, c = idx % MAXA;
  out[idx] = Cp[n * MAXA_P + c];
}

// ---------------------------------------------------------------------------
extern "C" void kernel_launch(void* const* d_in, const int* in_sizes, int n_in,
                              void* d_out, int out_size, void* d_ws, size_t ws_size,
                              hipStream_t stream)
{
  (void)in_sizes; (void)n_in; (void)out_size; (void)ws_size;
  // params flattened jax-style (sorted dict keys, list order):
  // 0 atom_latent.W, 1 atom_latent.b, 2 coord_out.W, 3 fc_atom.W, 4 fc_atom.b,
  // 5 final_ln.b, 6 final_ln.g, 7 lattice_out.W,
  // 8+10l: edge1.W, edge1.b, edge2.W, edge2.b, ln.b, ln.g, node1.W, node1.b, node2.W, node2.b
  // 68 node_emb.W, 69 node_emb.b, 70 time_emb(0), 71 atom_types, 72 frac,
  // 73 lattice, 74 node2graph, 75 edge_src, 76 edge_dst, 77 num_atoms
  auto F = [&](int i) { return (const float*)d_in[i]; };
  auto I = [&](int i) { return (const int*)d_in[i]; };

  const float* atomlat_W = F(0);
  const float* atomlat_b = F(1);
  const float* coord_W   = F(2);
  const float* fcatom_W  = F(3);
  const float* fcatom_b  = F(4);
  const float* fln_b     = F(5);
  const float* fln_g     = F(6);
  const float* latt_W    = F(7);
  const float* nemb_W    = F(68);
  const float* nemb_b    = F(69);
  const float* atom_t    = F(71);
  const float* frac      = F(72);
  const float* lattice   = F(73);
  const int*   n2g       = I(74);
  const int*   esrc      = I(75);
  const int*   edst      = I(76);

  // workspace carve (~119 MB)
  size_t off = 0;
  auto carve = [&](size_t bytes) {
    void* p = (char*)d_ws + off;
    off += (bytes + 255) & ~(size_t)255;
    return p;
  };
  _Float16* ws_dis = (_Float16*)carve((size_t)NEDGES * DIS_DIM * 2);
  float* ws_R     = (float*)carve((size_t)NEDGES * HID * 4);  // dis-proj -> ef1 (in place)
  float* ws_ef2   = (float*)carve((size_t)NEDGES * HID * 4);
  float* ws_h     = (float*)carve((size_t)NNODES * HID * 4);
  float* ws_hl    = (float*)carve((size_t)NNODES * HID * 4);
  float* ws_t0    = (float*)carve((size_t)NNODES * HID * 4);
  float* ws_P     = (float*)carve((size_t)NNODES * 1024 * 4); // [Psrc | Pdst]
  float* ws_Q     = (float*)carve((size_t)NGRAPH * HID * 4);
  float* ws_agg   = (float*)carve((size_t)NNODES * HID * 4);
  float* ws_ncat  = (float*)carve((size_t)NNODES * 1024 * 4);
  float* ws_nf1   = (float*)carve((size_t)NNODES * HID * 4);
  float* ws_lip   = (float*)carve((size_t)NGRAPH * 9 * 4);
  float* ws_gf    = (float*)carve((size_t)NGRAPH * HID * 4);
  float* ws_latm  = (float*)carve((size_t)NGRAPH * 9 * 4);
  float* ws_fcW   = (float*)carve((size_t)HID * MAXA_P * 4);  // padded fc_atom weight
  float* ws_fcb   = (float*)carve((size_t)MAXA_P * 4);
  float* ws_fcout = (float*)carve((size_t)NNODES * MAXA_P * 4);

  float* out_lat   = (float*)d_out;          // 32*9
  float* out_coord = out_lat + NGRAPH * 9;   // 768*3
  float* out_atom  = out_coord + NNODES * 3; // 768*100

  auto gemm_f32 = [&](const float* A, const float* W, const float* bias,
                      const float* resid, float* C, int M, int N, int K, int ldc, int act) {
    dim3 g(N / BN, M / BM);
    hipLaunchKernelGGL((wmma_gemm_kernel<0>), g, dim3(256), 0, stream,
                       (const void*)A, W, bias, resid, C, M, N, K, ldc, act);
  };
  auto gemm_f16 = [&](const _Float16* A, const float* W, const float* bias,
                      const float* resid, float* C, int M, int N, int K, int ldc, int act) {
    dim3 g(N / BN, M / BM);
    hipLaunchKernelGGL((wmma_gemm_kernel<1>), g, dim3(256), 0, stream,
                       (const void*)A, W, bias, resid, C, M, N, K, ldc, act);
  };

  // --- precompute edge geometry features ---
  k_dis  <<<(NEDGES * DIS_DIM) / 256, 256, 0, stream>>>(frac, esrc, edst, ws_dis);
  k_latip<<<2, 256, 0, stream>>>(lattice, ws_lip);

  // --- node embedding: h = atom_latent(node_emb(atom_types)) ---
  k_h0<<<(NNODES * HID) / 256, 256, 0, stream>>>(atom_t, nemb_W, nemb_b, ws_t0);
  gemm_f32(ws_t0, atomlat_W, atomlat_b, nullptr, ws_h, NNODES, HID, HID, HID, 0);

  // --- message-passing layers ---
  for (int l = 0; l < NLAYERS; ++l) {
    const float* W1  = F(8 + 10 * l + 0);
    const float* b1  = F(8 + 10 * l + 1);
    const float* W2  = F(8 + 10 * l + 2);
    const float* b2  = F(8 + 10 * l + 3);
    const float* lnb = F(8 + 10 * l + 4);
    const float* lng = F(8 + 10 * l + 5);
    const float* Wn1 = F(8 + 10 * l + 6);
    const float* bn1 = F(8 + 10 * l + 7);
    const float* Wn2 = F(8 + 10 * l + 8);
    const float* bn2 = F(8 + 10 * l + 9);

    k_ln<<<NNODES, 256, 0, stream>>>(ws_h, lng, lnb, ws_hl);

    // edge1 decomposition: node-level src/dst projections (packed ldc=1024)
    gemm_f32(ws_hl, W1,                     nullptr, nullptr, ws_P,       NNODES, HID, HID, 1024, 0);
    gemm_f32(ws_hl, W1 + (size_t)512 * HID, nullptr, nullptr, ws_P + 512, NNODES, HID, HID, 1024, 0);
    k_qlat<<<(NGRAPH * HID) / 256, 256, 0, stream>>>(ws_lip, W1 + (size_t)1024 * HID, ws_Q);
    // the only edge-level K=768 GEMM: dis @ W_dis
    gemm_f16(ws_dis, W1 + (size_t)1033 * HID, nullptr, nullptr, ws_R, NEDGES, HID, DIS_DIM, HID, 0);
    // combine + bias + silu -> ef1 (in place over ws_R)
    k_combine<<<(NEDGES * HID) / 256, 256, 0, stream>>>(ws_R, ws_P, ws_Q, b1, esrc, edst, n2g);
    // edge2 + silu
    gemm_f32(ws_R, W2, b2, nullptr, ws_ef2, NEDGES, HID, HID, HID, 1);
    // per-node mean over the 24 contiguous edges
    k_agg<<<NNODES, 256, 0, stream>>>(ws_ef2, ws_agg);
    // node MLP with residual
    k_concat<<<(NNODES * 1024) / 256, 256, 0, stream>>>(ws_hl, ws_agg, ws_ncat);
    gemm_f32(ws_ncat, Wn1, bn1, nullptr, ws_nf1, NNODES, HID, 1024, HID, 1);
    gemm_f32(ws_nf1,  Wn2, bn2, ws_h,    ws_h,   NNODES, HID, HID,  HID, 1);
  }

  // --- heads ---
  k_ln<<<NNODES, 256, 0, stream>>>(ws_h, fln_g, fln_b, ws_hl);        // hF
  k_coord<<<(NNODES * 3) / 256, 256, 0, stream>>>(ws_hl, coord_W, out_coord);
  k_gfeat<<<NGRAPH, 256, 0, stream>>>(ws_hl, ws_gf);
  k_lat1<<<2, 256, 0, stream>>>(ws_gf, latt_W, ws_latm);
  k_lat2<<<2, 256, 0, stream>>>(ws_latm, lattice, out_lat);
  // fc_atom via padded N=128 GEMM (keeps the hot GEMM kernel branch-free)
  k_padfc<<<(HID * MAXA_P) / 256, 256, 0, stream>>>(fcatom_W, fcatom_b, ws_fcW, ws_fcb);
  gemm_f32(ws_hl, ws_fcW, ws_fcb, nullptr, ws_fcout, NNODES, MAXA_P, HID, MAXA_P, 0);
  k_extract<<<(NNODES * MAXA) / 256, 256, 0, stream>>>(ws_fcout, out_atom);
}
